// LoRARelationalGATEncoder_24696061952627
// MI455X (gfx1250) — matrix operand order repaired and checked by reading, not compiled, not run on hardware
//
#include <hip/hip_runtime.h>
#include <hip/hip_bf16.h>
#include <math.h>

#define EMB   128
#define HID   128
#define OUTD  256
#define HEADS 4
#define RANK  8
#define NREL  8
#define HC1   (HEADS*HID)   // 512
#define D2    (HEADS*HID)   // 512
#define SCALE_F 1.0f
#define NEG_SLOPE 0.2f
#define LN_EPS 1e-5f

typedef __attribute__((ext_vector_type(16))) __bf16        v16bf;
typedef __attribute__((ext_vector_type(8)))  float          v8f;
typedef __attribute__((ext_vector_type(8)))  unsigned short ushort8;

union Frag { ushort8 h[2]; v16bf v; };

__device__ __forceinline__ unsigned short f2bf(float f) {
  unsigned int u = __float_as_uint(f);
  unsigned int r = (u + 0x7FFFu + ((u >> 16) & 1u)) >> 16;
  return (unsigned short)r;
}

__device__ __forceinline__ void atomicMaxF(float* addr, float val) {
  if (val >= 0.0f) atomicMax((int*)addr, __float_as_int(val));
  else             atomicMin((unsigned int*)addr, __float_as_uint(val));
}

// ---------------------------------------------------------------------------
// bf16 WMMA GEMM, register-blocked: C[M,N] = A[M,K] @ Bt[N,K]^T.
// One wave32 computes a 16x64 C tile: one A fragment is reused across 4 B
// fragments / 4 f32 accumulators per K-step (cuts A traffic 4x, load:wmma
// ratio 2.5:1). A,Bt bf16 row-major; C f32 row-major.
// Requires M%16==0, N%64==0, K%32==0.
// Fragments: A 16x32, B 32x16 with contiguous k (lane half h holds k = 16h..16h+15); f32 C 16x16.
// ---------------------------------------------------------------------------
__global__ void wmma_gemm_bf16(const unsigned short* __restrict__ A,
                               const unsigned short* __restrict__ Bt,
                               float* __restrict__ C, int M, int N, int K) {
  int wave = (int)((blockIdx.x * blockDim.x + threadIdx.x) >> 5);
  int lane = threadIdx.x & 31;
  int tilesN = N >> 6;                 // 64-wide N blocks
  int tM = (wave / tilesN) << 4;
  int tN = (wave % tilesN) << 6;
  if (tM >= M) return;
  int half = lane >> 4;    // 0 | 1
  int l16  = lane & 15;
  const unsigned short* arow = A  + (size_t)(tM + l16) * K;
  const unsigned short* brow = Bt + (size_t)(tN + l16) * K;   // + nb*16*K per sub-tile
  v8f acc0 = {}, acc1 = {}, acc2 = {}, acc3 = {};
  const size_t bstep = (size_t)16 * K;
  for (int kb = 0; kb < K; kb += 32) {
    Frag fa;
    // A: lane half -> k in {half*16 .. half*16+15}
    fa.h[0] = *reinterpret_cast<const ushort8*>(arow + kb + half * 16);
    fa.h[1] = *reinterpret_cast<const ushort8*>(arow + kb + half * 16 + 8);
    // B: lane half -> k in {half*16 .. half*16+15}, column = l16 (+16 per sub-tile)
    Frag fb0, fb1, fb2, fb3;
    const unsigned short* bk = brow + kb + half * 16;
    fb0.h[0] = *reinterpret_cast<const ushort8*>(bk);
    fb0.h[1] = *reinterpret_cast<const ushort8*>(bk + 8);
    fb1.h[0] = *reinterpret_cast<const ushort8*>(bk + bstep);
    fb1.h[1] = *reinterpret_cast<const ushort8*>(bk + bstep + 8);
    fb2.h[0] = *reinterpret_cast<const ushort8*>(bk + 2 * bstep);
    fb2.h[1] = *reinterpret_cast<const ushort8*>(bk + 2 * bstep + 8);
    fb3.h[0] = *reinterpret_cast<const ushort8*>(bk + 3 * bstep);
    fb3.h[1] = *reinterpret_cast<const ushort8*>(bk + 3 * bstep + 8);
    acc0 = __builtin_amdgcn_wmma_f32_16x16x32_bf16(false, fa.v, false, fb0.v, (short)0, acc0, false, false);
    acc1 = __builtin_amdgcn_wmma_f32_16x16x32_bf16(false, fa.v, false, fb1.v, (short)0, acc1, false, false);
    acc2 = __builtin_amdgcn_wmma_f32_16x16x32_bf16(false, fa.v, false, fb2.v, (short)0, acc2, false, false);
    acc3 = __builtin_amdgcn_wmma_f32_16x16x32_bf16(false, fa.v, false, fb3.v, (short)0, acc3, false, false);
  }
  // C: vgpr j, lane -> row tM + j + 8*half, col tN + nb*16 + l16
  float* crow = C + (size_t)(tM + 8 * half) * N + tN + l16;
#pragma unroll
  for (int j = 0; j < 8; ++j) {
    crow[(size_t)j * N]      = acc0[j];
    crow[(size_t)j * N + 16] = acc1[j];
    crow[(size_t)j * N + 32] = acc2[j];
    crow[(size_t)j * N + 48] = acc3[j];
  }
}

// ------------------------- small utility kernels ---------------------------
__global__ void k_f32_to_bf16(const float* __restrict__ s, unsigned short* __restrict__ d, int n) {
  for (int i = blockIdx.x * blockDim.x + threadIdx.x; i < n; i += gridDim.x * blockDim.x)
    d[i] = f2bf(s[i]);
}

__global__ void k_fill_f32(float* __restrict__ p, float v, int n) {
  for (int i = blockIdx.x * blockDim.x + threadIdx.x; i < n; i += gridDim.x * blockDim.x)
    p[i] = v;
}

// Btcat[rk][d] = B[r, d*RANK + k]   (bf16)
__global__ void k_bt_convert(const float* __restrict__ B, unsigned short* __restrict__ dst, int dDim) {
  int n = NREL * RANK * dDim;
  for (int i = blockIdx.x * blockDim.x + threadIdx.x; i < n; i += gridDim.x * blockDim.x) {
    int rk = i / dDim, d = i - rk * dDim;
    int r = rk >> 3, k = rk & 7;
    dst[i] = f2bf(B[(size_t)r * dDim * RANK + d * RANK + k]);
  }
}

// M1[rk*512 + hc] = sum_d A1[r, d*8+k] * W1[hc, d]   (d=128)
__global__ void k_M1(const float* __restrict__ A1, const float* __restrict__ W1, float* __restrict__ M1) {
  int idx = blockIdx.x * blockDim.x + threadIdx.x;
  if (idx >= 64 * HC1) return;
  int rk = idx >> 9, hc = idx & 511;
  int r = rk >> 3, k = rk & 7;
  const float* a = A1 + (size_t)r * EMB * RANK;
  const float* w = W1 + (size_t)hc * EMB;
  float acc = 0.f;
  for (int d = 0; d < EMB; ++d) acc += a[d * RANK + k] * w[d];
  M1[idx] = acc;
}

// M2[rk*256 + o] = sum_d A2[r, d*8+k] * W2[o, d]   (d=512)
__global__ void k_M2(const float* __restrict__ A2, const float* __restrict__ W2, float* __restrict__ M2) {
  int idx = blockIdx.x * blockDim.x + threadIdx.x;
  if (idx >= 64 * OUTD) return;
  int rk = idx >> 8, o = idx & 255;
  int r = rk >> 3, k = rk & 7;
  const float* a = A2 + (size_t)r * D2 * RANK;
  const float* w = W2 + (size_t)o * D2;
  float acc = 0.f;
  for (int d = 0; d < D2; ++d) acc += a[d * RANK + k] * w[d];
  M2[idx] = acc;
}

// Q1[rk*4+h] = sum_c M1[rk*512 + h*128 + c] * att_r1[h*128 + c]
__global__ void k_Q1(const float* __restrict__ M1, const float* __restrict__ attr, float* __restrict__ Q1) {
  int idx = blockIdx.x * blockDim.x + threadIdx.x;
  if (idx >= 64 * HEADS) return;
  int rk = idx >> 2, h = idx & 3;
  float acc = 0.f;
  for (int c = 0; c < HID; ++c) acc += M1[(size_t)rk * HC1 + h * HID + c] * attr[h * HID + c];
  Q1[idx] = acc;
}

// Q2[rk] = sum_c M2[rk*256 + c] * att_r2[c]
__global__ void k_Q2(const float* __restrict__ M2, const float* __restrict__ attr, float* __restrict__ Q2) {
  int idx = blockIdx.x * blockDim.x + threadIdx.x;
  if (idx >= 64) return;
  float acc = 0.f;
  for (int c = 0; c < OUTD; ++c) acc += M2[(size_t)idx * OUTD + c] * attr[c];
  Q2[idx] = acc;
}

// per-(node,head) attention dot products against z1
__global__ void k_node_attn1(const float* __restrict__ z1, const float* __restrict__ al,
                             const float* __restrict__ ar, float* __restrict__ sL,
                             float* __restrict__ sR, int N) {
  int idx = blockIdx.x * blockDim.x + threadIdx.x;
  if (idx >= N * HEADS) return;
  int n = idx >> 2, h = idx & 3;
  const float* zp = z1 + (size_t)n * HC1 + h * HID;
  const float* a = al + h * HID;
  const float* b = ar + h * HID;
  float x = 0.f, y = 0.f;
  for (int c = 0; c < HID; ++c) { x += zp[c] * a[c]; y += zp[c] * b[c]; }
  sL[idx] = x; sR[idx] = y;
}

__global__ void k_node_attn2(const float* __restrict__ z2, const float* __restrict__ al,
                             const float* __restrict__ ar, float* __restrict__ sL,
                             float* __restrict__ sR, int N) {
  int n = blockIdx.x * blockDim.x + threadIdx.x;
  if (n >= N) return;
  const float* zp = z2 + (size_t)n * OUTD;
  float x = 0.f, y = 0.f;
  for (int c = 0; c < OUTD; ++c) { x += zp[c] * al[c]; y += zp[c] * ar[c]; }
  sL[n] = x; sR[n] = y;
}

// ------------------------------- layer 1 edges -----------------------------
__global__ void k_edge_alpha1(const int* __restrict__ ej, const int* __restrict__ ei,
                              const int* __restrict__ et, const float* __restrict__ P1,
                              const float* __restrict__ sL, const float* __restrict__ sR,
                              const float* __restrict__ Q1, const float* __restrict__ relb,
                              float* __restrict__ alpha, float* __restrict__ segmax,
                              int E, int N) {
  int e = blockIdx.x * blockDim.x + threadIdx.x;
  if (e >= E) return;
  int j = ej[e], i = ei[e], r = et[e];
  float low[RANK];
#pragma unroll
  for (int k = 0; k < RANK; ++k) low[k] = P1[(size_t)j * 64 + r * RANK + k];
  size_t seg = (size_t)r * N + i;
#pragma unroll
  for (int h = 0; h < HEADS; ++h) {
    float a = sL[i * HEADS + h] + sR[j * HEADS + h] + relb[r * HEADS + h];
    float acc = 0.f;
#pragma unroll
    for (int k = 0; k < RANK; ++k) acc += low[k] * Q1[(r * RANK + k) * HEADS + h];
    a += SCALE_F * acc;
    a = a > 0.f ? a : NEG_SLOPE * a;
    alpha[(size_t)e * HEADS + h] = a;
    atomicMaxF(&segmax[seg * HEADS + h], a);
  }
}

__global__ void k_edge_expsum1(const int* __restrict__ ei, const int* __restrict__ et,
                               float* __restrict__ alpha, const float* __restrict__ segmax,
                               float* __restrict__ segsum, int E, int N) {
  int e = blockIdx.x * blockDim.x + threadIdx.x;
  if (e >= E) return;
  size_t seg = (size_t)et[e] * N + ei[e];
#pragma unroll
  for (int h = 0; h < HEADS; ++h) {
    float ex = expf(alpha[(size_t)e * HEADS + h] - segmax[seg * HEADS + h]);
    alpha[(size_t)e * HEADS + h] = ex;
    atomicAdd(&segsum[seg * HEADS + h], ex);
  }
}

// one 128-thread block per edge; each thread does 4 channels of 512
__global__ void k_edge_agg1(const int* __restrict__ ej, const int* __restrict__ ei,
                            const int* __restrict__ et, const float* __restrict__ P1,
                            const float* __restrict__ z1, const float* __restrict__ M1,
                            const float* __restrict__ alpha, const float* __restrict__ segsum,
                            float* __restrict__ agg, int N) {
  int e = blockIdx.x, t = threadIdx.x;
  int j = ej[e], i = ei[e], r = et[e];
  float low[RANK];
#pragma unroll
  for (int k = 0; k < RANK; ++k) low[k] = P1[(size_t)j * 64 + r * RANK + k];
  size_t seg = (size_t)r * N + i;
#pragma unroll
  for (int q = 0; q < 4; ++q) {
    int c = t + q * 128;
    int h = c >> 7;
    float w = alpha[(size_t)e * HEADS + h] / segsum[seg * HEADS + h];
    float zj = z1[(size_t)j * HC1 + c];
    float acc = 0.f;
#pragma unroll
    for (int k = 0; k < RANK; ++k) acc += low[k] * M1[(size_t)(r * RANK + k) * HC1 + c];
    zj += SCALE_F * acc;
    atomicAdd(&agg[(size_t)i * HC1 + c], zj * w);
  }
}

// x1b = bf16(elu(agg1 + bias1))
__global__ void k_final_x1(const float* __restrict__ agg, const float* __restrict__ bias,
                           unsigned short* __restrict__ x1b, int total) {
  for (int i = blockIdx.x * blockDim.x + threadIdx.x; i < total; i += gridDim.x * blockDim.x) {
    int c = i & (HC1 - 1);
    float v = agg[i] + bias[c];
    v = v > 0.f ? v : expm1f(v);
    x1b[i] = f2bf(v);
  }
}

// ------------------------------- layer 2 edges -----------------------------
__global__ void k_edge_alpha2(const int* __restrict__ ej, const int* __restrict__ ei,
                              const int* __restrict__ et, const float* __restrict__ P2,
                              const float* __restrict__ sL, const float* __restrict__ sR,
                              const float* __restrict__ Q2, const float* __restrict__ relb,
                              float* __restrict__ alpha, float* __restrict__ segmax,
                              int E, int N) {
  int e = blockIdx.x * blockDim.x + threadIdx.x;
  if (e >= E) return;
  int j = ej[e], i = ei[e], r = et[e];
  float acc = 0.f;
#pragma unroll
  for (int k = 0; k < RANK; ++k) acc += P2[(size_t)j * 64 + r * RANK + k] * Q2[r * RANK + k];
  float a = sL[i] + sR[j] + relb[r] + SCALE_F * acc;
  a = a > 0.f ? a : NEG_SLOPE * a;
  alpha[e] = a;
  atomicMaxF(&segmax[(size_t)r * N + i], a);
}

__global__ void k_edge_expsum2(const int* __restrict__ ei, const int* __restrict__ et,
                               float* __restrict__ alpha, const float* __restrict__ segmax,
                               float* __restrict__ segsum, int E, int N) {
  int e = blockIdx.x * blockDim.x + threadIdx.x;
  if (e >= E) return;
  size_t seg = (size_t)et[e] * N + ei[e];
  float ex = expf(alpha[e] - segmax[seg]);
  alpha[e] = ex;
  atomicAdd(&segsum[seg], ex);
}

// one 256-thread block per edge, one thread per output channel
__global__ void k_edge_agg2(const int* __restrict__ ej, const int* __restrict__ ei,
                            const int* __restrict__ et, const float* __restrict__ P2,
                            const float* __restrict__ z2, const float* __restrict__ M2,
                            const float* __restrict__ alpha, const float* __restrict__ segsum,
                            float* __restrict__ agg, int N) {
  int e = blockIdx.x, c = threadIdx.x;
  int j = ej[e], i = ei[e], r = et[e];
  float w = alpha[e] / segsum[(size_t)r * N + i];
  float acc = 0.f;
#pragma unroll
  for (int k = 0; k < RANK; ++k)
    acc += P2[(size_t)j * 64 + r * RANK + k] * M2[(size_t)(r * RANK + k) * OUTD + c];
  float zj = z2[(size_t)j * OUTD + c] + SCALE_F * acc;
  atomicAdd(&agg[(size_t)i * OUTD + c], zj * w);
}

// out = LN(agg2 + bias2 + res + res_b) * g + b     (one 256-thread block / node)
__global__ void k_final_ln(const float* __restrict__ agg, const float* __restrict__ res,
                           const float* __restrict__ bias2, const float* __restrict__ res_b,
                           const float* __restrict__ g, const float* __restrict__ b,
                           float* __restrict__ out) {
  int n = blockIdx.x, c = threadIdx.x;
  float v = agg[(size_t)n * OUTD + c] + bias2[c] + res[(size_t)n * OUTD + c] + res_b[c];
  __shared__ float red[OUTD];
  red[c] = v; __syncthreads();
  for (int s = OUTD / 2; s > 0; s >>= 1) { if (c < s) red[c] += red[c + s]; __syncthreads(); }
  float mu = red[0] / (float)OUTD;
  __syncthreads();
  float d = v - mu;
  red[c] = d * d; __syncthreads();
  for (int s = OUTD / 2; s > 0; s >>= 1) { if (c < s) red[c] += red[c + s]; __syncthreads(); }
  float var = red[0] / (float)OUTD;
  out[(size_t)n * OUTD + c] = d * rsqrtf(var + LN_EPS) * g[c] + b[c];
}

// ---------------------------------------------------------------------------
static void launch_gemm(const unsigned short* A, const unsigned short* Bt, float* C,
                        int M, int N, int K, hipStream_t s) {
  int waves = (M / 16) * (N / 64);         // 16x64 C tile per wave
  int blocks = (waves + 7) / 8;            // 8 waves (256 threads) per block
  wmma_gemm_bf16<<<blocks, 256, 0, s>>>(A, Bt, C, M, N, K);
}

extern "C" void kernel_launch(void* const* d_in, const int* in_sizes, int n_in,
                              void* d_out, int out_size, void* d_ws, size_t ws_size,
                              hipStream_t stream) {
  const float* x0     = (const float*)d_in[0];
  const float* A1     = (const float*)d_in[1];
  const float* B1     = (const float*)d_in[2];
  const float* W1     = (const float*)d_in[3];
  const float* attl1  = (const float*)d_in[4];
  const float* attr1  = (const float*)d_in[5];
  const float* relb1  = (const float*)d_in[6];
  const float* bias1  = (const float*)d_in[7];
  const float* A2     = (const float*)d_in[8];
  const float* B2     = (const float*)d_in[9];
  const float* W2     = (const float*)d_in[10];
  const float* attl2  = (const float*)d_in[11];
  const float* attr2  = (const float*)d_in[12];
  const float* relb2  = (const float*)d_in[13];
  const float* bias2  = (const float*)d_in[14];
  const float* res_W  = (const float*)d_in[15];
  const float* res_b  = (const float*)d_in[16];
  const float* ln_g   = (const float*)d_in[17];
  const float* ln_b   = (const float*)d_in[18];
  const int*   eidx   = (const int*)d_in[19];   // [2,E]: row0=j, row1=i
  const int*   etype  = (const int*)d_in[20];
  float* out = (float*)d_out;

  const int N = in_sizes[0] / EMB;
  const int E = in_sizes[20];
  const int* ej = eidx;
  const int* ei = eidx + E;

  // ---- bump allocator on d_ws ----
  char* ws = (char*)d_ws;
  size_t off = 0;
  auto alloc = [&](size_t bytes) -> void* {
    off = (off + 255) & ~(size_t)255;
    void* p = ws + off;
    off += bytes;
    return p;
  };
  auto allocF = [&](size_t n) { return (float*)alloc(n * sizeof(float)); };
  auto allocH = [&](size_t n) { return (unsigned short*)alloc(n * sizeof(unsigned short)); };

  unsigned short* xb    = allocH((size_t)N * EMB);
  unsigned short* x1b   = allocH((size_t)N * HC1);
  unsigned short* W1b   = allocH((size_t)HC1 * EMB);
  unsigned short* W2b   = allocH((size_t)OUTD * D2);
  unsigned short* resWb = allocH((size_t)OUTD * EMB);
  unsigned short* B1tb  = allocH((size_t)64 * EMB);
  unsigned short* B2tb  = allocH((size_t)64 * D2);

  float* z1    = allocF((size_t)N * HC1);
  float* agg1  = allocF((size_t)N * HC1);
  float* P1    = allocF((size_t)N * 64);
  float* res   = allocF((size_t)N * OUTD);
  float* z2    = allocF((size_t)N * OUTD);
  float* agg2  = allocF((size_t)N * OUTD);
  float* P2    = allocF((size_t)N * 64);
  float* M1    = allocF((size_t)64 * HC1);
  float* M2    = allocF((size_t)64 * OUTD);
  float* Q1    = allocF(64 * HEADS);
  float* Q2    = allocF(64);
  float* sL1   = allocF((size_t)N * HEADS);
  float* sR1   = allocF((size_t)N * HEADS);
  float* sL2   = allocF(N);
  float* sR2   = allocF(N);
  float* alpha1  = allocF((size_t)E * HEADS);
  float* alpha2  = allocF(E);
  float* segmax1 = allocF((size_t)NREL * N * HEADS);
  float* segsum1 = allocF((size_t)NREL * N * HEADS);
  float* segmax2 = allocF((size_t)NREL * N);
  float* segsum2 = allocF((size_t)NREL * N);

  const int T = 256;
  auto gs = [&](int n) { return (n + T - 1) / T; };

  // ---- 0. bf16 conversions + tiny precomputes ----
  k_f32_to_bf16<<<gs(N * EMB),   T, 0, stream>>>(x0,    xb,    N * EMB);
  k_f32_to_bf16<<<gs(HC1 * EMB), T, 0, stream>>>(W1,    W1b,   HC1 * EMB);
  k_f32_to_bf16<<<gs(OUTD * D2), T, 0, stream>>>(W2,    W2b,   OUTD * D2);
  k_f32_to_bf16<<<gs(OUTD * EMB),T, 0, stream>>>(res_W, resWb, OUTD * EMB);
  k_bt_convert<<<gs(64 * EMB), T, 0, stream>>>(B1, B1tb, EMB);
  k_bt_convert<<<gs(64 * D2),  T, 0, stream>>>(B2, B2tb, D2);
  k_M1<<<gs(64 * HC1),  T, 0, stream>>>(A1, W1, M1);
  k_M2<<<gs(64 * OUTD), T, 0, stream>>>(A2, W2, M2);
  k_Q1<<<1, 256, 0, stream>>>(M1, attr1, Q1);
  k_Q2<<<1, 64,  0, stream>>>(M2, attr2, Q2);

  // ---- 1. layer-1 dense (WMMA) ----
  launch_gemm(xb, W1b,   z1,  N, HC1,  EMB, stream);   // z1  = x @ W1^T
  launch_gemm(xb, B1tb,  P1,  N, 64,   EMB, stream);   // P1  = x @ B1cat
  launch_gemm(xb, resWb, res, N, OUTD, EMB, stream);   // res = x @ res_W^T
  k_node_attn1<<<gs(N * HEADS), T, 0, stream>>>(z1, attl1, attr1, sL1, sR1, N);

  // ---- 2. layer-1 edge softmax + aggregation ----
  hipMemsetAsync(segsum1, 0, (size_t)NREL * N * HEADS * sizeof(float), stream);
  hipMemsetAsync(agg1,    0, (size_t)N * HC1 * sizeof(float), stream);
  k_fill_f32<<<gs(NREL * N * HEADS), T, 0, stream>>>(segmax1, -INFINITY, NREL * N * HEADS);
  k_edge_alpha1 <<<gs(E), T, 0, stream>>>(ej, ei, etype, P1, sL1, sR1, Q1, relb1, alpha1, segmax1, E, N);
  k_edge_expsum1<<<gs(E), T, 0, stream>>>(ei, etype, alpha1, segmax1, segsum1, E, N);
  k_edge_agg1   <<<E, 128, 0, stream>>>(ej, ei, etype, P1, z1, M1, alpha1, segsum1, agg1, N);
  k_final_x1<<<gs(N * HC1), T, 0, stream>>>(agg1, bias1, x1b, N * HC1);

  // ---- 3. layer-2 dense (WMMA) ----
  launch_gemm(x1b, W2b,  z2, N, OUTD, D2, stream);     // z2 = x1 @ W2^T
  launch_gemm(x1b, B2tb, P2, N, 64,   D2, stream);     // P2 = x1 @ B2cat
  k_node_attn2<<<gs(N), T, 0, stream>>>(z2, attl2, attr2, sL2, sR2, N);

  // ---- 4. layer-2 edge softmax + aggregation ----
  hipMemsetAsync(segsum2, 0, (size_t)NREL * N * sizeof(float), stream);
  hipMemsetAsync(agg2,    0, (size_t)N * OUTD * sizeof(float), stream);
  k_fill_f32<<<gs(NREL * N), T, 0, stream>>>(segmax2, -INFINITY, NREL * N);
  k_edge_alpha2 <<<gs(E), T, 0, stream>>>(ej, ei, etype, P2, sL2, sR2, Q2, relb2, alpha2, segmax2, E, N);
  k_edge_expsum2<<<gs(E), T, 0, stream>>>(ei, etype, alpha2, segmax2, segsum2, E, N);
  k_edge_agg2   <<<E, OUTD, 0, stream>>>(ej, ei, etype, P2, z2, M2, alpha2, segsum2, agg2, N);

  // ---- 5. residual + layernorm ----
  k_final_ln<<<N, OUTD, 0, stream>>>(agg2, res, bias2, res_b, ln_g, ln_b, out);

  (void)n_in; (void)out_size; (void)ws_size;
}
